// TopKSAE_14654428414271
// MI455X (gfx1250) — matrix operand order repaired and checked by reading, not compiled, not run on hardware
//
#include <hip/hip_runtime.h>
#include <hip/hip_bf16.h>

// Problem constants (match reference)
#define DIM   768
#define FEAT  16384
#define TOPK  64
#define BATCH 4096

typedef __attribute__((ext_vector_type(16))) __bf16 v16bf;
typedef __attribute__((ext_vector_type(8)))  float  v8f;

// ---------------------------------------------------------------------------
// fp32 -> bf16 (round-to-nearest-even) as a raw u16 pattern
__device__ __forceinline__ unsigned short f32_to_bf16(float f) {
    unsigned u = __float_as_uint(f);
    u = (u + 0x7FFFu + ((u >> 16) & 1u)) >> 16;
    return (unsigned short)u;
}

// ---------------------------------------------------------------------------
// Convert x to bf16 with b_dec subtracted, written in plain row-major layout:
// per 32-wide K chunk, K-groups of 8 kept in order {g0,g1,g2,g3} so that the GEMM
// loads 16 contiguous bf16 per lane (lanes 0-15 K{0-15}, lanes 16-31 K{16-31}),
// matching the contiguous-K B-fragment loads.
__global__ void __launch_bounds__(256)
sae_convert_x(const float* __restrict__ x, const float* __restrict__ b_dec,
              unsigned short* __restrict__ xb) {
    int i = blockIdx.x * 256 + threadIdx.x;           // destination index
    if (i >= BATCH * DIM) return;
    int row = i / DIM;
    int jj  = i - row * DIM;
    int chunk = jj >> 5;
    int j     = jj & 31;
    int g     = j >> 3;
    int sg    = g;                                    // no group reorder
    int src_col = (chunk << 5) | (sg << 3) | (j & 7);
    float v = x[(size_t)row * DIM + src_col] - b_dec[src_col];
    xb[i] = f32_to_bf16(v);
}

// Straight conversion of W_enc (B-fragment per-lane layout is contiguous K)
__global__ void __launch_bounds__(256)
sae_convert_w(const float* __restrict__ w, unsigned short* __restrict__ wb) {
    int i = blockIdx.x * 256 + threadIdx.x;
    if (i >= FEAT * DIM) return;
    wb[i] = f32_to_bf16(w[i]);
}

// ---------------------------------------------------------------------------
// Transpose W_dec (DIM x FEAT) -> W_decT (FEAT x DIM) so decode gathers are
// coalesced.  32x32 LDS tile, padded stride.
__global__ void __launch_bounds__(256)
sae_transpose_wdec(const float* __restrict__ wdec, float* __restrict__ wdecT) {
    __shared__ float t[32][33];
    int x0 = blockIdx.x * 32;   // feature dim
    int y0 = blockIdx.y * 32;   // d dim
    int tx = threadIdx.x, ty = threadIdx.y;
    #pragma unroll
    for (int j = ty; j < 32; j += 8)
        t[j][tx] = wdec[(size_t)(y0 + j) * FEAT + (x0 + tx)];
    __syncthreads();
    #pragma unroll
    for (int j = ty; j < 32; j += 8)
        wdecT[(size_t)(x0 + j) * DIM + (y0 + tx)] = t[tx][j];
}

// ---------------------------------------------------------------------------
// Encode GEMM: pre[B,F] = Xb[B,768] * W_encb[F,768]^T + b_enc
// Workgroup = 256 threads = 8 waves; 64x64 output tile.
// Wave (wid&3, wid>>2) owns a 16x32 sub-tile => 2x v_wmma_f32_16x16x32_bf16
// per 32-deep K step, 24 K steps.
__global__ void __launch_bounds__(256)
sae_encode_gemm(const unsigned short* __restrict__ Xb,
                const unsigned short* __restrict__ Wb,
                const float* __restrict__ b_enc,
                float* __restrict__ pre) {
    const int lane = threadIdx.x & 31;
    const int wid  = threadIdx.x >> 5;
    const int m0 = (wid & 3) << 4;                 // 0,16,32,48
    const int n0 = (wid >> 2) << 5;                // 0,32
    const int mrow = blockIdx.y * 64 + m0 + (lane & 15);
    const int frow = blockIdx.x * 64 + n0 + (lane & 15);
    const int kh   = (lane >> 4) << 4;             // 0 or 16 (per-lane K half)

    const unsigned short* ap  = Xb + (size_t)mrow * DIM + kh;
    const unsigned short* bp0 = Wb + (size_t)frow * DIM + kh;
    const unsigned short* bp1 = bp0 + (size_t)16 * DIM;

    v8f c0 = {}; v8f c1 = {};
    #pragma unroll 4
    for (int k0 = 0; k0 < DIM; k0 += 32) {
        // prefetch next K tile into cache hierarchy
        if (k0 + 32 < DIM) {
            __builtin_prefetch(ap  + k0 + 32, 0, 3);
            __builtin_prefetch(bp0 + k0 + 32, 0, 3);
            __builtin_prefetch(bp1 + k0 + 32, 0, 3);
        }
        v16bf a  = *reinterpret_cast<const v16bf*>(ap  + k0);
        v16bf b0 = *reinterpret_cast<const v16bf*>(bp0 + k0);
        v16bf b1 = *reinterpret_cast<const v16bf*>(bp1 + k0);
        c0 = __builtin_amdgcn_wmma_f32_16x16x32_bf16(false, a, false, b0,
                                                     (short)0, c0, false, false);
        c1 = __builtin_amdgcn_wmma_f32_16x16x32_bf16(false, a, false, b1,
                                                     (short)0, c1, false, false);
    }

    // C layout: lanes 0-15 -> M=r, lanes 16-31 -> M=r+8; N = lane&15
    const int col0 = blockIdx.x * 64 + n0 + (lane & 15);
    const int col1 = col0 + 16;
    const float be0 = b_enc[col0];
    const float be1 = b_enc[col1];
    const int rowbase = blockIdx.y * 64 + m0 + ((lane >> 4) << 3);
    float* o = pre + (size_t)rowbase * FEAT;
    #pragma unroll
    for (int r = 0; r < 8; ++r) {
        o[(size_t)r * FEAT + col0] = c0[r] + be0;
        o[(size_t)r * FEAT + col1] = c1[r] + be1;
    }
}

// ---------------------------------------------------------------------------
// Exact top-64 per row via 4-pass 8-bit radix select, entire row cached in
// LDS (64KB -- CDNA5 WGP has 320KB).  Keys are order-preserving u32 images
// of the floats; selection emits (value, index) compacted per row.
__global__ void __launch_bounds__(256)
sae_topk(const float* __restrict__ pre,
         float* __restrict__ vals, int* __restrict__ idxs) {
    extern __shared__ unsigned char smem[];
    unsigned* keys = (unsigned*)smem;                 // FEAT entries (64KB)
    unsigned* hist = keys + FEAT;                     // 256 entries
    int*      ctl  = (int*)(hist + 256);              // [0]=prefix [1]=want [2]=nout

    const int tid = threadIdx.x;
    const int row = blockIdx.x;
    const float* p = pre + (size_t)row * FEAT;

    // load row, map to sortable keys
    for (int i = tid; i < FEAT; i += 256) {
        unsigned b = __float_as_uint(p[i]);
        keys[i] = (b & 0x80000000u) ? ~b : (b | 0x80000000u);
    }
    if (tid == 0) { ctl[0] = 0; ctl[1] = TOPK; ctl[2] = 0; }
    __syncthreads();

    unsigned prefix = 0, maskhi = 0;
    #pragma unroll
    for (int pass = 0; pass < 4; ++pass) {
        const int pos = 24 - 8 * pass;
        hist[tid] = 0;
        __syncthreads();
        for (int i = tid; i < FEAT; i += 256) {
            unsigned k = keys[i];
            if ((k & maskhi) == prefix)
                atomicAdd(&hist[(k >> pos) & 255u], 1u);
        }
        __syncthreads();
        if (tid == 0) {
            int want = ctl[1];
            int cum = 0, chosen = 0;
            for (int b = 255; b >= 0; --b) {
                cum += (int)hist[b];
                if (cum >= want) { chosen = b; want -= (cum - (int)hist[b]); break; }
            }
            ctl[0] = (int)(prefix | ((unsigned)chosen << pos));
            ctl[1] = want;
        }
        __syncthreads();
        prefix = (unsigned)ctl[0];
        maskhi |= (0xFFu << pos);
    }
    const unsigned thresh = prefix;   // exact key of the 64th-largest element

    // strictly-greater elements first (guaranteed in)
    for (int i = tid; i < FEAT; i += 256) {
        unsigned k = keys[i];
        if (k > thresh) {
            int s = atomicAdd(&ctl[2], 1);
            unsigned b = (k & 0x80000000u) ? (k & 0x7FFFFFFFu) : ~k;
            vals[(size_t)row * TOPK + s] = __uint_as_float(b);
            idxs[(size_t)row * TOPK + s] = i;
        }
    }
    __syncthreads();
    // then ties, until the 64 slots are full
    for (int i = tid; i < FEAT; i += 256) {
        if (keys[i] == thresh) {
            int s = atomicAdd(&ctl[2], 1);
            if (s < TOPK) {
                unsigned b = (thresh & 0x80000000u) ? (thresh & 0x7FFFFFFFu) : ~thresh;
                vals[(size_t)row * TOPK + s] = __uint_as_float(b);
                idxs[(size_t)row * TOPK + s] = i;
            }
        }
    }
}

// ---------------------------------------------------------------------------
// Sparse decode: x_hat[b,:] = sum_k val * W_decT[idx,:] + b_dec
// One WG per row; W_decT (48MB) stays resident in the 192MB L2.
__global__ void __launch_bounds__(256)
sae_decode(const float* __restrict__ wdecT,
           const float* __restrict__ vals, const int* __restrict__ idxs,
           const float* __restrict__ b_dec, float* __restrict__ out) {
    __shared__ float sv[TOPK];
    __shared__ int   si[TOPK];
    const int tid = threadIdx.x;
    const int row = blockIdx.x;
    if (tid < TOPK) {
        sv[tid] = vals[(size_t)row * TOPK + tid];
        si[tid] = idxs[(size_t)row * TOPK + tid];
    }
    __syncthreads();
    float a0 = 0.f, a1 = 0.f, a2 = 0.f;
    #pragma unroll 4
    for (int k = 0; k < TOPK; ++k) {
        const float v = sv[k];
        const float* w = wdecT + (size_t)si[k] * DIM;
        a0 = fmaf(v, w[tid],       a0);
        a1 = fmaf(v, w[tid + 256], a1);
        a2 = fmaf(v, w[tid + 512], a2);
    }
    float* o = out + (size_t)row * DIM;
    o[tid]       = a0 + b_dec[tid];
    o[tid + 256] = a1 + b_dec[tid + 256];
    o[tid + 512] = a2 + b_dec[tid + 512];
}

// ---------------------------------------------------------------------------
extern "C" void kernel_launch(void* const* d_in, const int* in_sizes, int n_in,
                              void* d_out, int out_size, void* d_ws, size_t ws_size,
                              hipStream_t stream) {
    const float* x     = (const float*)d_in[0];   // [B, D]
    const float* W_enc = (const float*)d_in[1];   // [F, D]
    const float* b_enc = (const float*)d_in[2];   // [F]
    const float* W_dec = (const float*)d_in[3];   // [D, F]
    const float* b_dec = (const float*)d_in[4];   // [D]
    float* out = (float*)d_out;

    char* ws = (char*)d_ws;
    size_t off = 0;
    unsigned short* xb  = (unsigned short*)(ws + off); off += (size_t)BATCH * DIM * 2;
    unsigned short* wb  = (unsigned short*)(ws + off); off += (size_t)FEAT  * DIM * 2;
    float* wdT  = (float*)(ws + off); off += (size_t)FEAT * DIM * 4;
    float* pre  = (float*)(ws + off); off += (size_t)BATCH * FEAT * 4;
    float* vals = (float*)(ws + off); off += (size_t)BATCH * TOPK * 4;
    int*   idx  = (int*)  (ws + off); off += (size_t)BATCH * TOPK * 4;

    sae_convert_x<<<(BATCH * DIM + 255) / 256, 256, 0, stream>>>(x, b_dec, xb);
    sae_convert_w<<<(FEAT * DIM + 255) / 256, 256, 0, stream>>>(W_enc, wb);
    sae_transpose_wdec<<<dim3(FEAT / 32, DIM / 32), dim3(32, 8), 0, stream>>>(W_dec, wdT);
    sae_encode_gemm<<<dim3(FEAT / 64, BATCH / 64), 256, 0, stream>>>(xb, wb, b_enc, pre);

    size_t topk_lds = (size_t)(FEAT + 256) * 4 + 16;
    sae_topk<<<BATCH, 256, topk_lds, stream>>>(pre, vals, idx);
    sae_decode<<<BATCH, 256, 0, stream>>>(wdT, vals, idx, b_dec, out);
}
